// S4Block_18803366821990
// MI455X (gfx1250) — matrix operand order repaired and checked by reading, not compiled, not run on hardware
//
#include <hip/hip_runtime.h>
#include <hip/hip_bf16.h>
#include <math.h>

#define B_  32
#define H_  256
#define L_  2048
#define N_  64
#define F_  32
#define NF  4096   // FFT length = 2L

#define BT_PITCH 264   // bf16 elements per LDS row: 256 + 8 pad -> 132-dword stride (bank-conflict-free)
#define FT_PITCH 40    // 32 + 8 pad

typedef __attribute__((ext_vector_type(16))) __bf16 v16bf;
typedef __attribute__((ext_vector_type(8)))  __bf16 v8bf;   // 128-bit
typedef __attribute__((ext_vector_type(8)))  float  v8f;

__device__ __forceinline__ v16bf combine16(v8bf lo, v8bf hi) {
    return __builtin_shufflevector(lo, hi, 0, 1, 2, 3, 4, 5, 6, 7,
                                           8, 9, 10, 11, 12, 13, 14, 15);
}

__device__ __forceinline__ float gelu_tanh(float x) {
    const float k0 = 0.7978845608028654f; // sqrt(2/pi)
    float x3 = x * x * x;
    return 0.5f * x * (1.0f + tanhf(k0 * (x + 0.044715f * x3)));
}

// In-LDS iterative radix-2 DIT complex FFT, length 4096, 256 threads.
// Forward only (sign -1); inverse via conjugate trick by caller.
__device__ void fft4096_inplace(float* re, float* im, int tid) {
    for (int i = tid; i < NF; i += 256) {
        int r = (int)(__brev((unsigned)i) >> 20);
        if (r > i) {
            float tr = re[i]; re[i] = re[r]; re[r] = tr;
            float ti = im[i]; im[i] = im[r]; im[r] = ti;
        }
    }
    __syncthreads();
    for (int len = 2; len <= NF; len <<= 1) {
        int half = len >> 1;
        float ang0 = -3.14159265358979323846f / (float)half;
        for (int j = tid; j < NF / 2; j += 256) {
            int blk = j / half, k = j - blk * half;
            int i0 = blk * len + k, i1 = i0 + half;
            float s, c;
            __sincosf(ang0 * (float)k, &s, &c);
            float xr = re[i1], xi = im[i1];
            float tr = c * xr - s * xi, ti = c * xi + s * xr;
            float ur = re[i0], ui = im[i0];
            re[i0] = ur + tr; im[i0] = ui + ti;
            re[i1] = ur - tr; im[i1] = ui - ti;
        }
        __syncthreads();
    }
}

// ---- Kernel 1: tb[b,g] = sum_h t[b,h]*Wt[g,h] + bt[g] ----
__global__ void k_time_linear(const float* __restrict__ t, const float* __restrict__ Wt,
                              const float* __restrict__ bt, float* __restrict__ tb) {
    int b = blockIdx.x, g = threadIdx.x;
    const float* tv = t + b * H_;
    const float* wr = Wt + g * H_;
    float acc = bt[g];
    for (int j = 0; j < H_; ++j) acc += tv[j] * wr[j];
    tb[b * H_ + g] = acc;
}

// ---- Kernel 2: fp32 weights -> bf16 copies for WMMA ----
__global__ void k_cvt_weights(const float* __restrict__ Wout, const float* __restrict__ W1,
                              const float* __restrict__ W2, const float* __restrict__ Wf,
                              __bf16* __restrict__ Woutb, __bf16* __restrict__ W1b,
                              __bf16* __restrict__ W2b, __bf16* __restrict__ Wfb) {
    int i = blockIdx.x * 256 + threadIdx.x;
    if (i < H_ * H_) {
        Woutb[i] = (__bf16)Wout[i];
        W1b[i]   = (__bf16)W1[i];
        W2b[i]   = (__bf16)W2[i];
    }
    if (i < H_ * F_) Wfb[i] = (__bf16)Wf[i];
}

// ---- Kernel 3: LayerNorm over H of xin = x + tb  -> z ----
__global__ void k_layernorm(const float* __restrict__ x, const float* __restrict__ tb,
                            const float* __restrict__ lng, const float* __restrict__ lnb,
                            float* __restrict__ z) {
    int b = blockIdx.y;
    int l = blockIdx.x * 256 + threadIdx.x;
    size_t base = ((size_t)b * H_) * L_ + l;
    const float* tbb = tb + b * H_;
    float s = 0.f, ss = 0.f;
    for (int h = 0; h < H_; ++h) {
        float v = x[base + (size_t)h * L_] + tbb[h];
        s += v; ss += v * v;
    }
    float mu  = s * (1.0f / H_);
    float var = ss * (1.0f / H_) - mu * mu;
    float inv = rsqrtf(var + 1e-5f);
    for (int h = 0; h < H_; ++h) {
        float v = x[base + (size_t)h * L_] + tbb[h];
        z[base + (size_t)h * L_] = (v - mu) * inv * lng[h] + lnb[h];
    }
}

// ---- Kernel 4: closed-form S4D kernel k[h,0:4096] + FFT -> spectrum Kf[h,:] ----
__global__ void k_ssm_kernel_fft(const float* __restrict__ log_dt, const float* __restrict__ logA_real,
                                 const float* __restrict__ A_imag, const float* __restrict__ C_re,
                                 const float* __restrict__ C_im, float2* __restrict__ Kf) {
    __shared__ float re[NF], im[NF];
    __shared__ float m_er[N_], m_ei[N_], c0r[N_], c0i[N_], c1r[N_], c1i[N_];
    int h = blockIdx.x, tid = threadIdx.x;
    if (tid < N_) {
        int n = tid;
        float dtv = __expf(log_dt[h]);
        float arr = __expf(logA_real[h * N_ + n]);   // A = (-arr, ai)
        float ai  = A_imag[h * N_ + n];
        float er = -dtv * arr, ei = dtv * ai;        // dtA
        float em = __expf(er);
        float sn, cs; __sincosf(ei, &sn, &cs);
        float Er = em * cs - 1.0f, Ei = em * sn;     // exp(dtA)-1
        float den = arr * arr + ai * ai;             // |A|^2
        float Fr = (-Er * arr + Ei * ai) / den;      // (exp(dtA)-1)/A
        float Fi = (-Ei * arr - Er * ai) / den;
        float cr0 = C_re[h * N_ + n],            ci0 = C_im[h * N_ + n];
        float cr1 = C_re[H_ * N_ + h * N_ + n],  ci1 = C_im[H_ * N_ + h * N_ + n];
        m_er[n] = er; m_ei[n] = ei;
        c0r[n] = cr0 * Fr - ci0 * Fi; c0i[n] = cr0 * Fi + ci0 * Fr;
        c1r[n] = cr1 * Fr - ci1 * Fi; c1i[n] = cr1 * Fi + ci1 * Fr;
    }
    __syncthreads();
    for (int l = tid; l < NF; l += 256) {
        int isRev = (l >= L_);
        float larg = isRev ? (float)(2 * L_ - 1 - l) : (float)l;  // k[2048+j] = K1[2047-j]
        const float* crp = isRev ? c1r : c0r;
        const float* cip = isRev ? c1i : c0i;
        float sum = 0.f;
        for (int n = 0; n < N_; ++n) {
            float mag = __expf(m_er[n] * larg);
            float sn, cs; __sincosf(m_ei[n] * larg, &sn, &cs);
            sum += mag * (crp[n] * cs - cip[n] * sn);             // Re(Cd * e^{dtA*l})
        }
        re[l] = 2.0f * sum; im[l] = 0.f;
    }
    __syncthreads();
    fft4096_inplace(re, im, tid);
    for (int i = tid; i < NF; i += 256)
        Kf[(size_t)h * NF + i] = make_float2(re[i], im[i]);
}

// ---- Kernel 5: per-(b,h) FFT convolution + D skip + GELU -> u (bf16) ----
__global__ void k_fft_conv(const float* __restrict__ z, const float2* __restrict__ Kf,
                           const float* __restrict__ D, __bf16* __restrict__ u) {
    __shared__ float re[NF], im[NF];
    int h = blockIdx.x, b = blockIdx.y, tid = threadIdx.x;
    size_t row = ((size_t)b * H_ + h) * L_;
    for (int i = tid; i < L_; i += 256) { re[i] = z[row + i]; im[i] = 0.f; }
    for (int i = L_ + tid; i < NF; i += 256) { re[i] = 0.f; im[i] = 0.f; }
    __syncthreads();
    fft4096_inplace(re, im, tid);                  // Z = FFT(pad(z))
    for (int i = tid; i < NF; i += 256) {          // conj(Z * Kf) -> inverse via fwd FFT
        float2 kf = Kf[(size_t)h * NF + i];
        float zr = re[i], zi = im[i];
        float pr = zr * kf.x - zi * kf.y;
        float pi = zr * kf.y + zi * kf.x;
        re[i] = pr; im[i] = -pi;
    }
    __syncthreads();
    fft4096_inplace(re, im, tid);                  // Re(result)/NF == IFFT real part
    float dh = D[h];
    const float scale = 1.0f / (float)NF;
    for (int i = tid; i < L_; i += 256) {
        float y = re[i] * scale + z[row + i] * dh;
        u[row + i] = (__bf16)gelu_tanh(y);
    }
}

// ---- Kernel 6: WMMA GEMM  g = gate(Wout@u + bout + xin + Wf@feat + bf) ----
// One block covers the full M=256 for one N-tile of 16: B panel staged once in LDS,
// each wave computes two 16x16 M-tiles sharing the B fragment.
__global__ void k_gemm_mid(const __bf16* __restrict__ u, const __bf16* __restrict__ Woutb,
                           const __bf16* __restrict__ Wfb, const float* __restrict__ features,
                           const float* __restrict__ x, const float* __restrict__ tb,
                           const float* __restrict__ bout, const float* __restrict__ bfv,
                           __bf16* __restrict__ g) {
    __shared__ __bf16 bT[16 * BT_PITCH];   // u panel transposed: [n][K], K = 256
    __shared__ __bf16 fT[16 * FT_PITCH];   // feature panel transposed: [n][K], K = 32
    int tid = threadIdx.x;
    int n0  = blockIdx.x * 16;
    int b   = blockIdx.z;

    // Stage u panel: thread t loads K-row t, 32 contiguous bytes, transposed store.
    {
        const __bf16* src = &u[((size_t)b * H_ + tid) * L_ + n0];
        v8bf q0 = *(const v8bf*)(src);
        v8bf q1 = *(const v8bf*)(src + 8);
        #pragma unroll
        for (int n = 0; n < 8; ++n) {
            bT[n * BT_PITCH + tid]       = q0[n];
            bT[(n + 8) * BT_PITCH + tid] = q1[n];
        }
    }
    // Stage feature panel (fp32 -> bf16), threads 0..31 each take K-row t.
    if (tid < F_) {
        const float* src = &features[((size_t)b * F_ + tid) * L_ + n0];
        #pragma unroll
        for (int n = 0; n < 16; ++n) fT[n * FT_PITCH + tid] = (__bf16)src[n];
    }
    __syncthreads();

    int wave = tid >> 5, lane = tid & 31;
    int lidx = lane & 15, half = lane >> 4;
    int m0a = wave * 16, m0b = 128 + wave * 16;

    v8f accA = {}, accB = {};
    for (int kk = 0; kk < H_; kk += 32) {
        const __bf16* bp = &bT[lidx * BT_PITCH + kk + half * 8];
        v16bf bb = combine16(*(const v8bf*)bp, *(const v8bf*)(bp + 16));
        const __bf16* wa = &Woutb[(m0a + lidx) * H_ + kk + half * 8];
        v16bf aA = combine16(*(const v8bf*)wa, *(const v8bf*)(wa + 16));
        const __bf16* wb = &Woutb[(m0b + lidx) * H_ + kk + half * 8];
        v16bf aB = combine16(*(const v8bf*)wb, *(const v8bf*)(wb + 16));
        if (kk + 32 < H_)
            __builtin_prefetch(wa + 32, 0, 1);
        accA = __builtin_amdgcn_wmma_f32_16x16x32_bf16(false, aA, false, bb, (short)0, accA, false, false);
        accB = __builtin_amdgcn_wmma_f32_16x16x32_bf16(false, aB, false, bb, (short)0, accB, false, false);
    }
    {   // feature encoder: K = F_ = 32, one step
        const __bf16* fp = &fT[lidx * FT_PITCH + half * 8];
        v16bf bb = combine16(*(const v8bf*)fp, *(const v8bf*)(fp + 16));
        const __bf16* wa = &Wfb[(m0a + lidx) * F_ + half * 8];
        v16bf aA = combine16(*(const v8bf*)wa, *(const v8bf*)(wa + 16));
        const __bf16* wb = &Wfb[(m0b + lidx) * F_ + half * 8];
        v16bf aB = combine16(*(const v8bf*)wb, *(const v8bf*)(wb + 16));
        accA = __builtin_amdgcn_wmma_f32_16x16x32_bf16(false, aA, false, bb, (short)0, accA, false, false);
        accB = __builtin_amdgcn_wmma_f32_16x16x32_bf16(false, aB, false, bb, (short)0, accB, false, false);
    }

    int l = n0 + lidx;
    #pragma unroll
    for (int r = 0; r < 8; ++r) {                 // C layout: M = m0 + r + 8*half, N = lane&15
        int mA = m0a + r + half * 8;
        size_t idxA = ((size_t)b * H_ + mA) * L_ + l;
        float vA = accA[r] + bout[mA] + bfv[mA] + x[idxA] + tb[b * H_ + mA];
        g[idxA] = (__bf16)(tanhf(vA) * (1.0f / (1.0f + __expf(-vA))));
        int mB = m0b + r + half * 8;
        size_t idxB = ((size_t)b * H_ + mB) * L_ + l;
        float vB = accB[r] + bout[mB] + bfv[mB] + x[idxB] + tb[b * H_ + mB];
        g[idxB] = (__bf16)(tanhf(vB) * (1.0f / (1.0f + __expf(-vB))));
    }
}

// ---- Kernel 7: dual WMMA GEMM  o1 = W1@g + b1 + x ; o2 = W2@g + b2 ----
// 4 accumulators (2 weight matrices x 2 M-tiles) share one B fragment per K-step.
__global__ void k_gemm_out(const __bf16* __restrict__ g, const __bf16* __restrict__ W1b,
                           const __bf16* __restrict__ W2b, const float* __restrict__ x,
                           const float* __restrict__ b1, const float* __restrict__ b2,
                           float* __restrict__ o1, float* __restrict__ o2) {
    __shared__ __bf16 bT[16 * BT_PITCH];   // g panel transposed: [n][K], K = 256
    int tid = threadIdx.x;
    int n0  = blockIdx.x * 16;
    int b   = blockIdx.z;
    {
        const __bf16* src = &g[((size_t)b * H_ + tid) * L_ + n0];
        v8bf q0 = *(const v8bf*)(src);
        v8bf q1 = *(const v8bf*)(src + 8);
        #pragma unroll
        for (int n = 0; n < 8; ++n) {
            bT[n * BT_PITCH + tid]       = q0[n];
            bT[(n + 8) * BT_PITCH + tid] = q1[n];
        }
    }
    __syncthreads();

    int wave = tid >> 5, lane = tid & 31;
    int lidx = lane & 15, half = lane >> 4;
    int m0a = wave * 16, m0b = 128 + wave * 16;

    v8f acc1A = {}, acc1B = {}, acc2A = {}, acc2B = {};
    for (int kk = 0; kk < H_; kk += 32) {
        const __bf16* bp = &bT[lidx * BT_PITCH + kk + half * 8];
        v16bf bb = combine16(*(const v8bf*)bp, *(const v8bf*)(bp + 16));
        const __bf16* w1a = &W1b[(m0a + lidx) * H_ + kk + half * 8];
        v16bf a1A = combine16(*(const v8bf*)w1a, *(const v8bf*)(w1a + 16));
        const __bf16* w1bp = &W1b[(m0b + lidx) * H_ + kk + half * 8];
        v16bf a1B = combine16(*(const v8bf*)w1bp, *(const v8bf*)(w1bp + 16));
        const __bf16* w2a = &W2b[(m0a + lidx) * H_ + kk + half * 8];
        v16bf a2A = combine16(*(const v8bf*)w2a, *(const v8bf*)(w2a + 16));
        const __bf16* w2bp = &W2b[(m0b + lidx) * H_ + kk + half * 8];
        v16bf a2B = combine16(*(const v8bf*)w2bp, *(const v8bf*)(w2bp + 16));
        if (kk + 32 < H_)
            __builtin_prefetch(w1a + 32, 0, 1);
        acc1A = __builtin_amdgcn_wmma_f32_16x16x32_bf16(false, a1A, false, bb, (short)0, acc1A, false, false);
        acc1B = __builtin_amdgcn_wmma_f32_16x16x32_bf16(false, a1B, false, bb, (short)0, acc1B, false, false);
        acc2A = __builtin_amdgcn_wmma_f32_16x16x32_bf16(false, a2A, false, bb, (short)0, acc2A, false, false);
        acc2B = __builtin_amdgcn_wmma_f32_16x16x32_bf16(false, a2B, false, bb, (short)0, acc2B, false, false);
    }

    int l = n0 + lidx;
    #pragma unroll
    for (int r = 0; r < 8; ++r) {
        int mA = m0a + r + half * 8;
        size_t idxA = ((size_t)b * H_ + mA) * L_ + l;
        o1[idxA] = acc1A[r] + b1[mA] + x[idxA];
        o2[idxA] = acc2A[r] + b2[mA];
        int mB = m0b + r + half * 8;
        size_t idxB = ((size_t)b * H_ + mB) * L_ + l;
        o1[idxB] = acc1B[r] + b1[mB] + x[idxB];
        o2[idxB] = acc2B[r] + b2[mB];
    }
}

extern "C" void kernel_launch(void* const* d_in, const int* in_sizes, int n_in,
                              void* d_out, int out_size, void* d_ws, size_t ws_size,
                              hipStream_t stream) {
    const float* x     = (const float*)d_in[0];
    const float* t     = (const float*)d_in[1];
    const float* feat  = (const float*)d_in[2];
    const float* Wt    = (const float*)d_in[3];
    const float* bt    = (const float*)d_in[4];
    const float* lng   = (const float*)d_in[5];
    const float* lnb   = (const float*)d_in[6];
    const float* logdt = (const float*)d_in[7];
    const float* logAr = (const float*)d_in[8];
    const float* Aim   = (const float*)d_in[9];
    const float* Cre   = (const float*)d_in[10];
    const float* Cim   = (const float*)d_in[11];
    const float* Dv    = (const float*)d_in[12];
    const float* Wout  = (const float*)d_in[13];
    const float* bout  = (const float*)d_in[14];
    const float* W1    = (const float*)d_in[15];
    const float* b1    = (const float*)d_in[16];
    const float* W2    = (const float*)d_in[17];
    const float* b2    = (const float*)d_in[18];
    const float* Wf    = (const float*)d_in[19];
    const float* bfv   = (const float*)d_in[20];

    char* ws = (char*)d_ws;
    size_t off = 0;
    float*  z     = (float*)(ws + off);  off += (size_t)B_ * H_ * L_ * 4;   // 64 MB
    float2* Kf    = (float2*)(ws + off); off += (size_t)H_ * NF * 8;        //  8 MB
    __bf16* u     = (__bf16*)(ws + off); off += (size_t)B_ * H_ * L_ * 2;   // 32 MB
    __bf16* g     = (__bf16*)(ws + off); off += (size_t)B_ * H_ * L_ * 2;   // 32 MB
    float*  tb    = (float*)(ws + off);  off += (size_t)B_ * H_ * 4;
    __bf16* Woutb = (__bf16*)(ws + off); off += (size_t)H_ * H_ * 2;
    __bf16* W1b   = (__bf16*)(ws + off); off += (size_t)H_ * H_ * 2;
    __bf16* W2b   = (__bf16*)(ws + off); off += (size_t)H_ * H_ * 2;
    __bf16* Wfb   = (__bf16*)(ws + off); off += (size_t)H_ * F_ * 2;

    float* o1 = (float*)d_out;
    float* o2 = o1 + (size_t)B_ * H_ * L_;

    k_time_linear   <<<dim3(B_), dim3(H_), 0, stream>>>(t, Wt, bt, tb);
    k_cvt_weights   <<<dim3((H_ * H_ + 255) / 256), 256, 0, stream>>>(Wout, W1, W2, Wf,
                                                                      Woutb, W1b, W2b, Wfb);
    k_layernorm     <<<dim3(L_ / 256, B_), 256, 0, stream>>>(x, tb, lng, lnb, z);
    k_ssm_kernel_fft<<<dim3(H_), 256, 0, stream>>>(logdt, logAr, Aim, Cre, Cim, Kf);
    k_fft_conv      <<<dim3(H_, B_), 256, 0, stream>>>(z, Kf, Dv, u);
    k_gemm_mid      <<<dim3(L_ / 16, 1, B_), 256, 0, stream>>>(u, Woutb, Wfb, feat,
                                                               x, tb, bout, bfv, g);
    k_gemm_out      <<<dim3(L_ / 16, 1, B_), 256, 0, stream>>>(g, W1b, W2b, x, b1, b2,
                                                               o1, o2);
}
